// MambaLayer_3702261809501
// MI455X (gfx1250) — compile-verified, matrix operand order checked by reading, not run on hardware
//
#include <hip/hip_runtime.h>
#include <hip/hip_bf16.h>

// ---------------------------------------------------------------------------
// Bidirectional Mamba block for MI455X (gfx1250, wave32, WMMA).
//  - GEMMs: v_wmma_f32_16x16x32_f16, register-blocked (1 wave = 1 M-tile x
//    several N-tiles, A-fragment loaded once per K-step).
//  - Selective scan: chunked associative scan (3 passes, chunk=128) so the
//    serial depth is ~384 instead of 16384, with 32K-wave parallelism.
// ---------------------------------------------------------------------------

typedef __attribute__((ext_vector_type(16))) _Float16 v16h;
typedef __attribute__((ext_vector_type(8)))  _Float16 v8h;
typedef __attribute__((ext_vector_type(8)))  float    v8f;

#define DIM      128
#define D_INNER  256
#define D_STATE  16
#define DT_RANK  8
#define NB       2
#define LSEQ     16384            // 16*32*32
#define BL       (NB * LSEQ)      // 32768 rows
#define NDBL_PAD 48               // 40 padded to 3 N-tiles
#define CS       128              // scan chunk size
#define NCH      (LSEQ / CS)      // 128 chunks

// ---- fragment loaders -----------------------------------------------------
// 16-bit A/B operand: lane t -> row/col (t&15), K-halves (t>>4)*8 in v[0..7],
// K-halves 16+(t>>4)*8 in v[8..15].  p points at K = k0 + (t>>4)*8.
__device__ inline v16h frag_load(const _Float16* __restrict__ p) {
  v8h lo = *(const v8h*)(p);
  v8h hi = *(const v8h*)(p + 16);
  v16h r;
#pragma unroll
  for (int i = 0; i < 8; ++i) { r[i] = lo[i]; r[i + 8] = hi[i]; }
  return r;
}

__device__ inline v16h frag_load_sum(const _Float16* __restrict__ p0,
                                     const _Float16* __restrict__ p1) {
  v8h a0 = *(const v8h*)(p0);       v8h a1 = *(const v8h*)(p0 + 16);
  v8h b0 = *(const v8h*)(p1);       v8h b1 = *(const v8h*)(p1 + 16);
  v16h r;
#pragma unroll
  for (int i = 0; i < 8; ++i) {
    r[i]     = (_Float16)((float)a0[i] + (float)b0[i]);
    r[i + 8] = (_Float16)((float)a1[i] + (float)b1[i]);
  }
  return r;
}

__device__ inline float silu(float x) { return x / (1.0f + expf(-x)); }

// ---- weight conversion ----------------------------------------------------
__global__ void k_cvt_f16(const float* __restrict__ src, _Float16* __restrict__ dst, int n) {
  int i = blockIdx.x * 256 + threadIdx.x;
  if (i < n) dst[i] = (_Float16)src[i];
}

// x_proj (40 x 256) -> padded (48 x 256), pad rows zero
__global__ void k_cvt_xproj(const float* __restrict__ src, _Float16* __restrict__ dst) {
  int i = blockIdx.x * 256 + threadIdx.x;
  if (i >= NDBL_PAD * D_INNER) return;
  int r = i / D_INNER;
  dst[i] = (r < (DT_RANK + 2 * D_STATE)) ? (_Float16)src[i] : (_Float16)0.0f;
}

// ---- LayerNorm: one wave per token row ------------------------------------
__global__ void k_layernorm(const float* __restrict__ X,   // (B, 128, L)
                            const float* __restrict__ lnw,
                            const float* __restrict__ lnb,
                            _Float16* __restrict__ Xn) {   // (BL, 128) f16
  const int lane = threadIdx.x;           // 0..31
  const long row = blockIdx.x;            // 0..BL-1
  const int b = (int)(row / LSEQ);
  const int l = (int)(row % LSEQ);
  const float* xp = X + (long)b * DIM * LSEQ + l;
  float v[4], s = 0.f, s2 = 0.f;
#pragma unroll
  for (int j = 0; j < 4; ++j) {
    int c = lane + j * 32;
    v[j] = xp[(long)c * LSEQ];
    s += v[j]; s2 += v[j] * v[j];
  }
#pragma unroll
  for (int m = 1; m < 32; m <<= 1) { s += __shfl_xor(s, m); s2 += __shfl_xor(s2, m); }
  const float mu  = s * (1.0f / DIM);
  const float var = s2 * (1.0f / DIM) - mu * mu;
  const float inv = rsqrtf(var + 1e-5f);
#pragma unroll
  for (int j = 0; j < 4; ++j) {
    int c = lane + j * 32;
    Xn[row * DIM + c] = (_Float16)((v[j] - mu) * inv * lnw[c] + lnb[c]);
  }
}

// ---- in_proj GEMM: (BL x 128) @ (128 x 512) -> xz f16 ---------------------
// Wave = 1 M-tile x 8 N-tiles: A-fragment loaded once per K-step.
__global__ void k_gemm_inproj(const _Float16* __restrict__ Xn,  // BL x 128
                              const _Float16* __restrict__ W,   // 512 x 128 f16
                              _Float16* __restrict__ XZ) {      // BL x 512
  const int lane = threadIdx.x & 31;
  const int wid  = blockIdx.x * (blockDim.x >> 5) + (threadIdx.x >> 5);
  const int mtile = wid >> 2;                      // BL/16 tiles
  const int ng    = wid & 3;                       // 4 groups of 8 N-tiles
  const int idx = lane & 15, half = lane >> 4, kb = half * 8;
  const _Float16* arow = Xn + (long)(mtile * 16 + idx) * DIM;
  v8f acc[8] = {};
#pragma unroll
  for (int k0 = 0; k0 < DIM; k0 += 32) {
    v16h a = frag_load(arow + k0 + kb);
#pragma unroll
    for (int t = 0; t < 8; ++t) {
      const _Float16* brow = W + (long)((ng * 8 + t) * 16 + idx) * DIM;
      v16h b = frag_load(brow + k0 + kb);
      acc[t] = __builtin_amdgcn_wmma_f32_16x16x32_f16(false, a, false, b, (short)0, acc[t],
                                                      false, false);
    }
  }
#pragma unroll
  for (int t = 0; t < 8; ++t) {
    _Float16* out = XZ + (long)(mtile * 16) * (2 * D_INNER) + (ng * 8 + t) * 16 + idx;
#pragma unroll
    for (int r = 0; r < 8; ++r)
      out[(long)(r + 8 * half) * (2 * D_INNER)] = (_Float16)acc[t][r];
  }
}

// ---- depthwise causal conv(4) + SiLU, both directions ---------------------
__global__ void k_conv_silu(const _Float16* __restrict__ XZ,
                            const float* __restrict__ wF, const float* __restrict__ bF,
                            const float* __restrict__ wB, const float* __restrict__ bB,
                            _Float16* __restrict__ XsF, _Float16* __restrict__ XsB) {
  const int branch = blockIdx.y;
  const long gid = (long)blockIdx.x * blockDim.x + threadIdx.x;   // BL*256
  const int d  = (int)(gid % D_INNER);
  const long row = gid / D_INNER;            // b*L + i (branch space)
  const int b = (int)(row / LSEQ);
  const int i = (int)(row % LSEQ);
  const float* w    = branch ? wB : wF;
  const float* bias = branch ? bB : bF;
  float acc = bias[d];
#pragma unroll
  for (int j = 0; j < 4; ++j) {
    int s = i - 3 + j;
    if (s >= 0) {
      int lsrc = branch ? (LSEQ - 1 - s) : s;
      acc += w[d * 4 + j] * (float)XZ[((long)b * LSEQ + lsrc) * (2 * D_INNER) + d];
    }
  }
  (branch ? XsB : XsF)[row * D_INNER + d] = (_Float16)silu(acc);
}

// ---- x_proj GEMM: (BL x 256) @ (256 x 48pad) -> dbl f32 -------------------
// Wave = 1 M-tile x all 3 N-tiles.
__global__ void k_gemm_xproj(const _Float16* __restrict__ Xs,   // BL x 256
                             const _Float16* __restrict__ Wxp,  // 48 x 256 f16
                             float* __restrict__ DBL) {         // BL x 48
  const int lane = threadIdx.x & 31;
  const int mtile = blockIdx.x * (blockDim.x >> 5) + (threadIdx.x >> 5);
  const int idx = lane & 15, half = lane >> 4, kb = half * 8;
  const _Float16* arow = Xs + (long)(mtile * 16 + idx) * D_INNER;
  v8f acc[3] = {};
#pragma unroll
  for (int k0 = 0; k0 < D_INNER; k0 += 32) {
    v16h a = frag_load(arow + k0 + kb);
#pragma unroll
    for (int t = 0; t < 3; ++t) {
      const _Float16* brow = Wxp + (long)(t * 16 + idx) * D_INNER;
      v16h b = frag_load(brow + k0 + kb);
      acc[t] = __builtin_amdgcn_wmma_f32_16x16x32_f16(false, a, false, b, (short)0, acc[t],
                                                      false, false);
    }
  }
#pragma unroll
  for (int t = 0; t < 3; ++t) {
    float* out = DBL + (long)(mtile * 16) * NDBL_PAD + t * 16 + idx;
#pragma unroll
    for (int r = 0; r < 8; ++r) out[(long)(r + 8 * half) * NDBL_PAD] = acc[t][r];
  }
}

// ---- dt_proj (rank 8) + softplus ------------------------------------------
__global__ void k_dtproj(const float* __restrict__ dblF, const float* __restrict__ dblB,
                         const float* __restrict__ wF, const float* __restrict__ biF,
                         const float* __restrict__ wB, const float* __restrict__ biB,
                         _Float16* __restrict__ dtF, _Float16* __restrict__ dtB) {
  const int branch = blockIdx.y;
  const long gid = (long)blockIdx.x * blockDim.x + threadIdx.x;   // BL*256
  const int d = (int)(gid % D_INNER);
  const long row = gid / D_INNER;
  const float* dbl  = branch ? dblB : dblF;
  const float* w    = branch ? wB : wF;
  const float* bias = branch ? biB : biF;
  float acc = bias[d];
#pragma unroll
  for (int r = 0; r < DT_RANK; ++r) acc += dbl[row * NDBL_PAD + r] * w[d * DT_RANK + r];
  float sp = (acc > 20.0f) ? acc : log1pf(expf(acc));
  (branch ? dtB : dtF)[row * D_INNER + d] = (_Float16)sp;
}

// ===========================================================================
// Chunked selective scan (associative): h_i = a_i h_{i-1} + u_i
//   pass 1: per chunk, decay product P and local end-state S (h_in = 0)
//   pass 2: tiny serial combine over chunks -> S becomes incoming state H
//   pass 3: re-run chunk with correct h_in, contract with C, gate, store y
// Wave layout: lanes 0-15 = states of channel d0, lanes 16-31 = channel d1.
// P/S layout: [((b*NCH + c)*D_INNER + d)*D_STATE + n]
// ===========================================================================
__global__ void k_scan_chunks(const _Float16* __restrict__ dt,
                              const _Float16* __restrict__ xs,
                              const float*    __restrict__ dbl,
                              const float*    __restrict__ A_log,
                              float* __restrict__ P, float* __restrict__ S) {
  const int lane = threadIdx.x;
  const int widx = blockIdx.x;                  // (b, dpair, chunk)
  const int c     = widx % NCH;
  const int rest  = widx / NCH;
  const int b     = rest / (D_INNER / 2);
  const int dpair = rest % (D_INNER / 2);
  const int g = lane >> 4, n = lane & 15;
  const int d = dpair * 2 + g;
  const float A = -expf(A_log[d * D_STATE + n]);
  const long base = (long)b * LSEQ + (long)c * CS;
  float h = 0.0f, pr = 1.0f;
  for (int i = 0; i < CS; ++i) {
    const long row = base + i;
    const float dtv = (float)dt[row * D_INNER + d];
    const float xv  = (float)xs[row * D_INNER + d];
    const float Bv  = dbl[row * NDBL_PAD + DT_RANK + n];
    const float a   = expf(dtv * A);
    h = a * h + (dtv * xv) * Bv;
    pr *= a;
  }
  const long idx = ((long)(b * NCH + c) * D_INNER + d) * D_STATE + n;
  P[idx] = pr;
  S[idx] = h;
}

__global__ void k_scan_combine(const float* __restrict__ P, float* __restrict__ S) {
  const int tid = blockIdx.x * blockDim.x + threadIdx.x;     // NB*D_INNER*D_STATE
  if (tid >= NB * D_INNER * D_STATE) return;
  const int b  = tid / (D_INNER * D_STATE);
  const int dn = tid % (D_INNER * D_STATE);                  // d*16 + n
  float H = 0.0f;
  for (int c = 0; c < NCH; ++c) {
    const long idx = (long)(b * NCH + c) * (D_INNER * D_STATE) + dn;
    const float p = P[idx];
    const float s = S[idx];
    S[idx] = H;                 // incoming state for chunk c
    H = p * H + s;
  }
}

__global__ void k_scan_apply(const _Float16* __restrict__ dt,
                             const _Float16* __restrict__ xs,
                             const float*    __restrict__ dbl,
                             const _Float16* __restrict__ XZ,  // z = cols [256,512)
                             const float*    __restrict__ A_log,
                             const float*    __restrict__ Dsk,
                             const float*    __restrict__ Hin, // == S after combine
                             _Float16* __restrict__ Y,
                             int branch) {
  const int lane = threadIdx.x;
  const int widx = blockIdx.x;
  const int c     = widx % NCH;
  const int rest  = widx / NCH;
  const int b     = rest / (D_INNER / 2);
  const int dpair = rest % (D_INNER / 2);
  const int g = lane >> 4, n = lane & 15;
  const int d = dpair * 2 + g;
  const float A  = -expf(A_log[d * D_STATE + n]);
  const float Dv = Dsk[d];
  const long bbase = (long)b * LSEQ;
  const long base  = bbase + (long)c * CS;
  float h = Hin[((long)(b * NCH + c) * D_INNER + d) * D_STATE + n];
  for (int i = 0; i < CS; ++i) {
    const long row = base + i;
    const float dtv = (float)dt[row * D_INNER + d];
    const float xv  = (float)xs[row * D_INNER + d];
    const float Bv  = dbl[row * NDBL_PAD + DT_RANK + n];
    const float Cv  = dbl[row * NDBL_PAD + DT_RANK + D_STATE + n];
    h = expf(dtv * A) * h + (dtv * xv) * Bv;
    float p = h * Cv;
    p += __shfl_xor(p, 1);
    p += __shfl_xor(p, 2);
    p += __shfl_xor(p, 4);
    p += __shfl_xor(p, 8);          // masks < 16 stay inside each half-wave
    if (n == 0) {
      const int iseq = (int)(row - bbase);
      const int lz = branch ? (LSEQ - 1 - iseq) : iseq;
      const float zv = (float)XZ[(bbase + lz) * (2 * D_INNER) + D_INNER + d];
      Y[row * D_INNER + d] = (_Float16)((p + xv * Dv) * silu(zv));
    }
  }
}

// ---- out_proj GEMM with fused y_f + reverse(y_b), transposed f32 store ----
// Wave = 1 M-tile x all 8 N-tiles: fused A-fragment built once per K-step.
__global__ void k_gemm_outproj(const _Float16* __restrict__ Yf,   // (BL,256) fwd space
                               const _Float16* __restrict__ Yb,   // (BL,256) rev space
                               const _Float16* __restrict__ Wout, // 128 x 256 f16
                               float* __restrict__ Out) {         // (B,128,L)
  const int lane = threadIdx.x & 31;
  const int mtile = blockIdx.x * (blockDim.x >> 5) + (threadIdx.x >> 5);
  const int idx = lane & 15, half = lane >> 4, kb = half * 8;
  const int rowF = mtile * 16 + idx;
  const int bb = rowF / LSEQ, ll = rowF % LSEQ;
  const long rowB = (long)bb * LSEQ + (LSEQ - 1 - ll);
  const _Float16* af = Yf + (long)rowF * D_INNER;
  const _Float16* ab = Yb + rowB * D_INNER;
  v8f acc[8] = {};
#pragma unroll
  for (int k0 = 0; k0 < D_INNER; k0 += 32) {
    v16h a = frag_load_sum(af + k0 + kb, ab + k0 + kb);
#pragma unroll
    for (int t = 0; t < 8; ++t) {
      const _Float16* brow = Wout + (long)(t * 16 + idx) * D_INNER;
      v16h b = frag_load(brow + k0 + kb);
      acc[t] = __builtin_amdgcn_wmma_f32_16x16x32_f16(false, a, false, b, (short)0, acc[t],
                                                      false, false);
    }
  }
#pragma unroll
  for (int t = 0; t < 8; ++t) {
#pragma unroll
    for (int r = 0; r < 8; ++r) {
      const int m = r + 8 * half;
      const int row = mtile * 16 + m;
      const int ob = row / LSEQ, ol = row % LSEQ;
      Out[(long)ob * DIM * LSEQ + (long)(t * 16 + idx) * LSEQ + ol] = acc[t][r];
    }
  }
}

// ---------------------------------------------------------------------------
extern "C" void kernel_launch(void* const* d_in, const int* in_sizes, int n_in,
                              void* d_out, int out_size, void* d_ws, size_t ws_size,
                              hipStream_t stream) {
  const float* x        = (const float*)d_in[0];
  const float* ln_w     = (const float*)d_in[1];
  const float* ln_b     = (const float*)d_in[2];
  const float* in_proj  = (const float*)d_in[3];
  const float* out_proj = (const float*)d_in[4];
  const float* conv_w_f = (const float*)d_in[5];
  const float* conv_b_f = (const float*)d_in[6];
  const float* xprj_f   = (const float*)d_in[7];
  const float* dtw_f    = (const float*)d_in[8];
  const float* dtb_f    = (const float*)d_in[9];
  const float* Alog_f   = (const float*)d_in[10];
  const float* Dsk_f    = (const float*)d_in[11];
  const float* conv_w_b = (const float*)d_in[12];
  const float* conv_b_b = (const float*)d_in[13];
  const float* xprj_b   = (const float*)d_in[14];
  const float* dtw_b    = (const float*)d_in[15];
  const float* dtb_b    = (const float*)d_in[16];
  const float* Alog_b   = (const float*)d_in[17];
  const float* Dsk_b    = (const float*)d_in[18];
  float* out = (float*)d_out;

  // ---- workspace bump allocator (256B aligned) ----
  char* ws = (char*)d_ws;
  size_t off = 0;
  auto alloc = [&](size_t bytes) {
    void* p = ws + off;
    off += (bytes + 255) & ~(size_t)255;
    return p;
  };
  _Float16* Wq_h    = (_Float16*)alloc((size_t)2 * D_INNER * DIM * 2);   // 512x128
  _Float16* Wout_h  = (_Float16*)alloc((size_t)DIM * D_INNER * 2);       // 128x256
  _Float16* Wxp_f_h = (_Float16*)alloc((size_t)NDBL_PAD * D_INNER * 2);  // 48x256
  _Float16* Wxp_b_h = (_Float16*)alloc((size_t)NDBL_PAD * D_INNER * 2);
  _Float16* Xn_h    = (_Float16*)alloc((size_t)BL * DIM * 2);
  _Float16* XZ_h    = (_Float16*)alloc((size_t)BL * 2 * D_INNER * 2);
  _Float16* XsF_h   = (_Float16*)alloc((size_t)BL * D_INNER * 2);
  _Float16* XsB_h   = (_Float16*)alloc((size_t)BL * D_INNER * 2);
  float*    dblF    = (float*)   alloc((size_t)BL * NDBL_PAD * 4);
  float*    dblB    = (float*)   alloc((size_t)BL * NDBL_PAD * 4);
  _Float16* dtF_h   = (_Float16*)alloc((size_t)BL * D_INNER * 2);
  _Float16* dtB_h   = (_Float16*)alloc((size_t)BL * D_INNER * 2);
  _Float16* YF_h    = (_Float16*)alloc((size_t)BL * D_INNER * 2);
  _Float16* YB_h    = (_Float16*)alloc((size_t)BL * D_INNER * 2);
  const size_t PS_N = (size_t)NB * NCH * D_INNER * D_STATE;              // 1M floats
  float* PF = (float*)alloc(PS_N * 4);
  float* SF = (float*)alloc(PS_N * 4);
  float* PB = (float*)alloc(PS_N * 4);
  float* SB = (float*)alloc(PS_N * 4);
  (void)ws_size; (void)n_in; (void)in_sizes; (void)out_size;

  // 1) weights -> f16
  k_cvt_f16<<<(2 * D_INNER * DIM + 255) / 256, 256, 0, stream>>>(in_proj, Wq_h, 2 * D_INNER * DIM);
  k_cvt_f16<<<(DIM * D_INNER + 255) / 256, 256, 0, stream>>>(out_proj, Wout_h, DIM * D_INNER);
  k_cvt_xproj<<<(NDBL_PAD * D_INNER + 255) / 256, 256, 0, stream>>>(xprj_f, Wxp_f_h);
  k_cvt_xproj<<<(NDBL_PAD * D_INNER + 255) / 256, 256, 0, stream>>>(xprj_b, Wxp_b_h);

  // 2) LayerNorm (wave per token)
  k_layernorm<<<BL, 32, 0, stream>>>(x, ln_w, ln_b, Xn_h);

  // 3) in_proj GEMM: 2048 M-tiles x 4 N-groups, 4 waves/block
  k_gemm_inproj<<<(BL / 16) * 4 / 4, 128, 0, stream>>>(Xn_h, Wq_h, XZ_h);

  // 4) conv + SiLU (both directions)
  {
    dim3 grid((unsigned)((long)BL * D_INNER / 256), 2);
    k_conv_silu<<<grid, 256, 0, stream>>>(XZ_h, conv_w_f, conv_b_f, conv_w_b, conv_b_b,
                                          XsF_h, XsB_h);
  }

  // 5) x_proj GEMM: wave per M-tile (3 N-tiles each)
  k_gemm_xproj<<<(BL / 16) / 4, 128, 0, stream>>>(XsF_h, Wxp_f_h, dblF);
  k_gemm_xproj<<<(BL / 16) / 4, 128, 0, stream>>>(XsB_h, Wxp_b_h, dblB);

  // 6) dt_proj + softplus
  {
    dim3 grid((unsigned)((long)BL * D_INNER / 256), 2);
    k_dtproj<<<grid, 256, 0, stream>>>(dblF, dblB, dtw_f, dtb_f, dtw_b, dtb_b, dtF_h, dtB_h);
  }

  // 7) chunked selective scan per branch
  const int scan_blocks = NB * (D_INNER / 2) * NCH;        // 32768 waves
  k_scan_chunks<<<scan_blocks, 32, 0, stream>>>(dtF_h, XsF_h, dblF, Alog_f, PF, SF);
  k_scan_combine<<<(NB * D_INNER * D_STATE + 255) / 256, 256, 0, stream>>>(PF, SF);
  k_scan_apply<<<scan_blocks, 32, 0, stream>>>(dtF_h, XsF_h, dblF, XZ_h, Alog_f, Dsk_f,
                                               SF, YF_h, 0);
  k_scan_chunks<<<scan_blocks, 32, 0, stream>>>(dtB_h, XsB_h, dblB, Alog_b, PB, SB);
  k_scan_combine<<<(NB * D_INNER * D_STATE + 255) / 256, 256, 0, stream>>>(PB, SB);
  k_scan_apply<<<scan_blocks, 32, 0, stream>>>(dtB_h, XsB_h, dblB, XZ_h, Alog_b, Dsk_b,
                                               SB, YB_h, 1);

  // 8) out_proj GEMM with fused y_f + reverse(y_b), transposed store
  k_gemm_outproj<<<(BL / 16) / 4, 128, 0, stream>>>(YF_h, YB_h, Wout_h, out);
}
